// Regressor1Stage_90048284328369
// MI455X (gfx1250) — matrix operand order repaired and checked by reading, not compiled, not run on hardware
//
#include <hip/hip_runtime.h>

// ---------------------------------------------------------------------------
// Fused Regressor1Stage for MI455X (gfx1250, wave32, WMMA).
//   grid = (W/32, H, B), block = 256 (8 waves)
//   Per block: 3-layer per-row MLP via v_wmma_f32_16x16x32_f16 (+bias+lrelu),
//   per-column argmax, class-conditioned 64->2 regression in f32, outputs.
// All intermediates live in LDS; x is read from HBM exactly once.
// ---------------------------------------------------------------------------

#define Cc   64      // channels
#define Hh   448     // rows (= conv groups)
#define Ww   608     // width
#define Kk   32      // classes per row
#define Bb   2       // batch
#define WT   32      // column tile per workgroup (608 = 19 * 32)
#define SW   72      // weight LDS row stride (halfs), 16B-aligned rows, padded
#define SA   72      // activation LDS row stride (halfs), N-major [w][c]
#define SX   65      // f32 x-copy LDS stride (floats), padded vs 64 banks
#define SCLS 33      // cls LDS stride (floats)

typedef __attribute__((ext_vector_type(16))) _Float16 v16h;
typedef __attribute__((ext_vector_type(8)))  _Float16 v8h;
typedef __attribute__((ext_vector_type(8)))  float    v8f;

__device__ __forceinline__ float lrelu(float v) { return v >= 0.f ? v : 0.01f * v; }

// A fragment (16x32 f16): rows m0..m0+15, K = kk..kk+31, from row-major [m][k].
// ISA 7.12.2: lane holds row m=lane&15; K = hsel*8+{0..7} and 16+hsel*8+{0..7}.
__device__ __forceinline__ v16h load_a_frag(const _Float16* p, int stride, int m0,
                                            int kk, int lane) {
  const int m    = m0 + (lane & 15);
  const int hsel = lane >> 4;
  const _Float16* base = p + m * stride + kk + hsel * 8;
  v8h lo = *(const v8h*)(base);
  v8h hi = *(const v8h*)(base + 16);
  return __builtin_shufflevector(lo, hi, 0,1,2,3,4,5,6,7,8,9,10,11,12,13,14,15);
}

// B fragment (32x16 f16): K = kk..kk+31, cols n0..n0+15, from N-major [n][k].
// Lane holds col n=lane&15; K = hsel*8+{0..7} and 16+hsel*8+{0..7} (matches A).
__device__ __forceinline__ v16h load_b_frag(const _Float16* p, int stride, int n0,
                                            int kk, int lane) {
  const int n    = n0 + (lane & 15);
  const int hsel = lane >> 4;
  const _Float16* base = p + n * stride + kk + hsel * 8;
  v8h lo = *(const v8h*)(base);
  v8h hi = *(const v8h*)(base + 16);
  return __builtin_shufflevector(lo, hi, 0,1,2,3,4,5,6,7,8,9,10,11,12,13,14,15);
}

// One 16x16 output tile of a [64 x WT] layer: K=64 (2 WMMA steps),
// epilogue = +bias, leaky-relu, f16 store transposed (N-major) for next layer.
__device__ __forceinline__ void gemm64x32_tile(const _Float16* wlds, const _Float16* alds,
                                               const float* biaslds, _Float16* outlds,
                                               int mt, int nt, int lane) {
  v8f acc = {};
#pragma unroll
  for (int kk = 0; kk < Cc; kk += 32) {
    v16h af = load_a_frag(wlds, SW, mt * 16, kk, lane);
    v16h bf = load_b_frag(alds, SA, nt * 16, kk, lane);
    acc = __builtin_amdgcn_wmma_f32_16x16x32_f16(false, af, false, bf,
                                                 (short)0, acc, false, false);
  }
  const int n  = nt * 16 + (lane & 15);
  const int mh = (lane >> 4) * 8;
#pragma unroll
  for (int r = 0; r < 8; ++r) {
    const int m = mt * 16 + r + mh;               // ISA C/D layout: M = r + 8*hsel
    outlds[n * SA + m] = (_Float16)lrelu(acc[r] + biaslds[m]);
  }
}

__global__ void __launch_bounds__(256)
regressor1stage_kernel(const float* __restrict__ x,
                       const float* __restrict__ W1, const float* __restrict__ b1,
                       const float* __restrict__ W2, const float* __restrict__ b2,
                       const float* __restrict__ W3, const float* __restrict__ b3,
                       const float* __restrict__ Wr, const float* __restrict__ br,
                       float* __restrict__ out) {
  __shared__ __align__(16) _Float16 act0[WT * SA];
  __shared__ __align__(16) _Float16 act1[WT * SA];
  __shared__ __align__(16) _Float16 wgt[Cc * SW];
  __shared__ float xorig[WT * SX];
  __shared__ float bias[Cc];
  __shared__ float cls[WT * SCLS];
  __shared__ int   clsidx[WT];
  __shared__ float red[WT * 8 * 2];

  const int tid  = threadIdx.x;
  const int lane = tid & 31;
  const int wave = tid >> 5;
  const int wt = blockIdx.x, h = blockIdx.y, b = blockIdx.z;
  const int w0 = wt * WT;
  const int HW = Hh * Ww;

  // ---- Stage 0: load x tile (f32 copy for cond-mul, f16 for WMMA) + W1/b1 ----
  {
    const int w  = tid & (WT - 1);                 // 0..31 (== lane)
    const int c0 = tid >> 5;                       // 0..7
    const int base0 = b * (Cc * HW) + h * Ww + w0 + w;
#pragma unroll
    for (int i = 0; i < 8; ++i) {
      const int c = c0 * 8 + i;
      const float v = x[base0 + c * HW];
      xorig[w * SX + c] = v;
      act0[w * SA + c]  = (_Float16)v;
    }
  }
  {
    const float* Wp = W1 + h * (Cc * Cc);
#pragma unroll
    for (int t = 0; t < (Cc * Cc) / 256; ++t) {
      const int i = tid + t * 256;
      wgt[(i >> 6) * SW + (i & 63)] = (_Float16)Wp[i];
    }
    if (tid < Cc) bias[tid] = b1[h * Cc + tid];
    if (tid < 128) __builtin_prefetch(W2 + h * (Cc * Cc) + tid * 32, 0, 1);
  }
  __syncthreads();

  // ---- Layer 1: act1 = lrelu(W1 @ act0 + b1) ----
  gemm64x32_tile(wgt, act0, bias, act1, wave & 3, wave >> 2, lane);
  __syncthreads();

  // ---- Load W2/b2 ----
  {
    const float* Wp = W2 + h * (Cc * Cc);
#pragma unroll
    for (int t = 0; t < (Cc * Cc) / 256; ++t) {
      const int i = tid + t * 256;
      wgt[(i >> 6) * SW + (i & 63)] = (_Float16)Wp[i];
    }
    if (tid < Cc) bias[tid] = b2[h * Cc + tid];
    if (tid < 64) __builtin_prefetch(W3 + h * (Kk * Cc) + tid * 32, 0, 1);
  }
  __syncthreads();

  // ---- Layer 2: act0 = lrelu(W2 @ act1 + b2) (x f16 copy is dead now) ----
  gemm64x32_tile(wgt, act1, bias, act0, wave & 3, wave >> 2, lane);
  __syncthreads();

  // ---- Load W3/b3 (32x64) ----
  {
    const float* Wp = W3 + h * (Kk * Cc);
#pragma unroll
    for (int t = 0; t < (Kk * Cc) / 256; ++t) {
      const int i = tid + t * 256;
      wgt[(i >> 6) * SW + (i & 63)] = (_Float16)Wp[i];
    }
    if (tid < Kk) bias[tid] = b3[h * Kk + tid];
  }
  __syncthreads();

  // ---- Layer 3: cls (32x32 f32), waves 0..3 (wave-uniform => EXEC all-ones) ----
  if (wave < 4) {
    const int mt = wave & 1, nt = wave >> 1;
    v8f acc = {};
#pragma unroll
    for (int kk = 0; kk < Cc; kk += 32) {
      v16h af = load_a_frag(wgt,  SW, mt * 16, kk, lane);
      v16h bf = load_b_frag(act0, SA, nt * 16, kk, lane);
      acc = __builtin_amdgcn_wmma_f32_16x16x32_f16(false, af, false, bf,
                                                   (short)0, acc, false, false);
    }
    const int n  = nt * 16 + (lane & 15);
    const int mh = (lane >> 4) * 8;
#pragma unroll
    for (int r = 0; r < 8; ++r) {
      const int m = mt * 16 + r + mh;
      cls[n * SCLS + m] = lrelu(acc[r] + bias[m]);
    }
  }
  __syncthreads();

  // ---- Argmax per column (strict '>' => first max, matches jnp.argmax) ----
  if (tid < WT) {
    const float* col = &cls[tid * SCLS];
    int best = 0; float bv = col[0];
#pragma unroll
    for (int k = 1; k < Kk; ++k) {
      const float v = col[k];
      if (v > bv) { bv = v; best = k; }
    }
    clsidx[tid] = h * Kk + best;                   // global class id
  }
  __syncthreads();

  // ---- Cond-mul: 8 threads/column, 8 channels each, f32 dot with Wr[class] ----
  {
    const int w = tid >> 3, part = tid & 7;
    const int g = clsidx[w];
    const float* wr = Wr + g * (Cc * 2) + part * 8 * 2;
    const float* xc = &xorig[w * SX + part * 8];
    float s0 = 0.f, s1 = 0.f;
#pragma unroll
    for (int c = 0; c < 8; ++c) {
      const float xv = xc[c];
      s0 += xv * wr[2 * c];
      s1 += xv * wr[2 * c + 1];
    }
    red[(w * 8 + part) * 2 + 0] = s0;
    red[(w * 8 + part) * 2 + 1] = s1;
  }
  __syncthreads();

  // ---- Reduce + finalize outputs: xpos then mask (concatenated flat) ----
  if (tid < WT) {
    const int w = tid;
    const int g = clsidx[w];
    float s0 = 0.f, s1 = 0.f;
#pragma unroll
    for (int p = 0; p < 8; ++p) {
      s0 += red[(w * 8 + p) * 2 + 0];
      s1 += red[(w * 8 + p) * 2 + 1];
    }
    const float r0 = lrelu(s0 + br[g * 2 + 0]);
    const float r1 = lrelu(s1 + br[g * 2 + 1]);
    const int idx = (b * Hh + h) * Ww + w0 + w;
    out[idx]               = ((float)g + r0) * (1.0f / Kk);  // xpos
    out[Bb * Hh * Ww + idx] = lrelu(r1);                     // mask
  }
}

extern "C" void kernel_launch(void* const* d_in, const int* in_sizes, int n_in,
                              void* d_out, int out_size, void* d_ws, size_t ws_size,
                              hipStream_t stream) {
  (void)in_sizes; (void)n_in; (void)out_size; (void)d_ws; (void)ws_size;
  const float* x  = (const float*)d_in[0];
  const float* W1 = (const float*)d_in[1];
  const float* b1 = (const float*)d_in[2];
  const float* W2 = (const float*)d_in[3];
  const float* b2 = (const float*)d_in[4];
  const float* W3 = (const float*)d_in[5];
  const float* b3 = (const float*)d_in[6];
  const float* Wr = (const float*)d_in[7];
  const float* br = (const float*)d_in[8];
  dim3 grid(Ww / WT, Hh, Bb);   // 19 x 448 x 2 workgroups
  dim3 block(256);              // 8 wave32s
  regressor1stage_kernel<<<grid, block, 0, stream>>>(x, W1, b1, W2, b2, W3, b3,
                                                     Wr, br, (float*)d_out);
}
